// mysnn_75866302317171
// MI455X (gfx1250) — matrix operand order repaired and checked by reading, not compiled, not run on hardware
//
#include <hip/hip_runtime.h>
#include <hip/hip_bf16.h>

// ---------------------------------------------------------------------------
// SNN forward on gfx1250 (MI455X).
//  - Per-layer pipeline: IIR(time) -> f16 pack -> WMMA GEMM -> spike scan.
//  - GEMM uses v_wmma_f32_16x16x32_f16 (wave32), one 16x16 tile per wave.
//  - Activations stored [B][Kpad][Tpad=304] f16; B-fragments are 16 strided
//    scalar loads per lane (stride T_PAD).
// ---------------------------------------------------------------------------

typedef __attribute__((ext_vector_type(16))) _Float16 v16h;
typedef __attribute__((ext_vector_type(8)))  _Float16 v8h;
typedef __attribute__((ext_vector_type(8)))  float    v8f;

#define T_LEN 300
#define T_PAD 304          // 19 * 16-col tiles; keeps 32B alignment (304*2 = 19*32)
#define BATCHN 64

// exp(-1/8), exp(-1/2) and derived IIR coefficients (reference constants)
#define EM_C   0.8824969025845955f
#define ES_C   0.6065306597126334f
#define A1_C   1.4890275622972289f     //  EM + ES
#define A2_C  (-0.5352614285189903f)   // -(EM*ES) = -exp(-0.625)
#define B_C    0.36795499049594947f    // (8/6)*(EM-ES)

// ---------------------------------------------------------------------------
// 1) dual-exp IIR over time, fused f32->f16 convert into padded tile buffer.
//    Covers the FULL padded extent [B][Kpad][T_PAD]: pad rows (f>=Fin) and pad
//    columns (t>=300) are written as zeros so WMMA reads are always valid.
// ---------------------------------------------------------------------------
__global__ void iir_to_f16(const float* __restrict__ src,   // [B][Fin][300]
                           const float* __restrict__ a1v,
                           const float* __restrict__ a2v,
                           const float* __restrict__ bv,
                           _Float16* __restrict__ xh,       // [B][Kpad][T_PAD]
                           int Fin, int Kpad)
{
    int idx = blockIdx.x * blockDim.x + threadIdx.x;
    int total = BATCHN * Kpad;
    if (idx >= total) return;
    int f = idx % Kpad;
    int b = idx / Kpad;
    _Float16* dst = xh + ((size_t)b * Kpad + f) * T_PAD;
    if (f >= Fin) {
        for (int t = 0; t < T_PAD; ++t) dst[t] = (_Float16)0.0f;
        return;
    }
    const float* s = src + ((size_t)b * Fin + f) * T_LEN;
    float a1 = a1v[f], a2 = a2v[f], bb = bv[f];
    float y1 = 0.0f, y2 = 0.0f;
    for (int t = 0; t < T_LEN; ++t) {
        float y = fmaf(a1, y1, fmaf(a2, y2, bb * s[t]));
        dst[t] = (_Float16)y;
        y2 = y1; y1 = y;
    }
    dst[300] = (_Float16)0.0f; dst[301] = (_Float16)0.0f;
    dst[302] = (_Float16)0.0f; dst[303] = (_Float16)0.0f;
}

// ---------------------------------------------------------------------------
// 2) pack weights f32 [Fout][Fin] -> zero-padded f16 [Mpad][Kpad]
// ---------------------------------------------------------------------------
__global__ void pack_w(const float* __restrict__ W, _Float16* __restrict__ Wh,
                       int Fout, int Fin, int Mpad, int Kpad)
{
    int idx = blockIdx.x * blockDim.x + threadIdx.x;
    int total = Mpad * Kpad;
    if (idx >= total) return;
    int i = idx % Kpad;
    int o = idx / Kpad;
    float v = (o < Fout && i < Fin) ? W[(size_t)o * Fin + i] : 0.0f;
    Wh[(size_t)o * Kpad + i] = (_Float16)v;
}

// ---------------------------------------------------------------------------
// 3) WMMA GEMM: cur[b][o][t] = sum_i Wh[o][i] * xh[b][i][t] + bias[o]
//    One wave per 16(M=o) x 16(N=t) tile; K loop in steps of 32.
//    A fragment (16x32 f16): lane m = l&15, hi = l>>4,
//      halves 0..7  -> K = k0 + 8*hi + j        (contiguous 16B)
//      halves 8..15 -> K = k0 + 16 + 8*hi + j-8 (contiguous 16B)
//    B fragment (32x16 f16): lane n = l&15 -> N = t0+n, halves j -> same K as A
//      -> 16 strided scalar loads per lane from the [f][t] activation layout.
//    D (16x16 f32): VGPR r -> row r + 8*hi, lane&15 -> col.
// ---------------------------------------------------------------------------
__global__ void __launch_bounds__(32)
gemm_wmma(const _Float16* __restrict__ xh,   // [B][Kpad][T_PAD]
          const _Float16* __restrict__ Wh,   // [Mpad][Kpad]
          const float* __restrict__ bias,    // [Fout]
          float* __restrict__ cur,           // [B][Fout][300]
          int Fout, int Kpad)
{
    const int lane = threadIdx.x;           // 0..31 (wave32)
    const int t0   = blockIdx.x * 16;       // N tile (time)
    const int o0   = blockIdx.y * 16;       // M tile (out features)
    const int b    = blockIdx.z;
    const int m16  = lane & 15;
    const int hi   = lane >> 4;

    v8f c = {};
    const _Float16* wrow = Wh + (size_t)(o0 + m16) * Kpad;
    const _Float16* xcol = xh + (size_t)b * Kpad * T_PAD + t0 + m16;

    for (int k0 = 0; k0 < Kpad; k0 += 32) {
        union { v16h v; v8h h[2]; } A;
        A.h[0] = *(const v8h*)(wrow + k0 + 8 * hi);
        A.h[1] = *(const v8h*)(wrow + k0 + 16 + 8 * hi);
        v16h Bf;
        for (int s = 0; s < 16; ++s)
            Bf[s] = xcol[(size_t)(k0 + (s < 8 ? 8 * hi + s : 16 + 8 * hi + s - 8)) * T_PAD];
        // (neg_a, A, neg_b, B, c_mod, C, reuse_a, reuse_b)
        c = __builtin_amdgcn_wmma_f32_16x16x32_f16(false, A.v, false, Bf,
                                                   (short)0, c, false, false);
    }

    const int t = t0 + m16;
    if (t < T_LEN) {
#pragma unroll
        for (int r = 0; r < 8; ++r) {
            int o = o0 + r + 8 * hi;
            if (o < Fout)
                cur[((size_t)b * Fout + o) * T_LEN + t] = c[r] + bias[o];
        }
    }
}

// ---------------------------------------------------------------------------
// 4) spiking neuron with decaying reset (membrane_filter == False)
// ---------------------------------------------------------------------------
__global__ void spike_scan(const float* __restrict__ cur,  // [B][Fout][300]
                           float* __restrict__ spk,        // [B][Fout][300]
                           int Fout)
{
    int idx = blockIdx.x * blockDim.x + threadIdx.x;
    int total = BATCHN * Fout;
    if (idx >= total) return;
    const float* c = cur + (size_t)idx * T_LEN;
    float* s = spk + (size_t)idx * T_LEN;
    float reset = 0.0f;
    for (int t = 0; t < T_LEN; ++t) {
        float v = c[t] + reset;
        float sp = (v >= 1.0f) ? 1.0f : 0.0f;
        reset = reset * EM_C - sp;   // reset*RESET_DECAY - s*RESET_V
        s[t] = sp;
    }
}

// ---------------------------------------------------------------------------
// 5) final fixed-coefficient output filter (f32 in, f32 out)
// ---------------------------------------------------------------------------
__global__ void out_filter(const float* __restrict__ s4,   // [B][300][300]
                           float* __restrict__ filt)
{
    int idx = blockIdx.x * blockDim.x + threadIdx.x;
    int total = BATCHN * 300;
    if (idx >= total) return;
    const float* x = s4 + (size_t)idx * T_LEN;
    float* y = filt + (size_t)idx * T_LEN;
    float y1 = 0.0f, y2 = 0.0f;
    for (int t = 0; t < T_LEN; ++t) {
        float yy = fmaf(A1_C, y1, fmaf(A2_C, y2, B_C * x[t]));
        y[t] = yy;
        y2 = y1; y1 = yy;
    }
}

// ---------------------------------------------------------------------------
// host launcher
// ---------------------------------------------------------------------------
extern "C" void kernel_launch(void* const* d_in, const int* in_sizes, int n_in,
                              void* d_out, int out_size, void* d_ws, size_t ws_size,
                              hipStream_t stream)
{
    (void)in_sizes; (void)n_in; (void)out_size; (void)ws_size;

    // layer dims: Fin, Fout, Kpad (Fin->mult 32), Mpad (Fout->mult 16)
    const int Fin[4]  = {300, 500, 200, 500};
    const int Fout[4] = {500, 200, 500, 300};
    const int Kpad[4] = {320, 512, 224, 512};
    const int Mpad[4] = {512, 208, 512, 304};
    // input dict indices: a1, a2, b, W, bias per layer (after inputs at 0)
    const int base[4] = {1, 6, 11, 16};

    // workspace layout (all 256B aligned; total ~97.3 MB)
    char* ws = (char*)d_ws;
    _Float16* xh  = (_Float16*)(ws);                       // 64*512*304 f16
    _Float16* Wh  = (_Float16*)(ws + 19922944);            // 512*512   f16
    float*    cur = (float*)   (ws + 20447232);            // 64*500*300 f32
    float*    spk = (float*)   (ws + 58847232);            // 64*500*300 f32

    const float* s_prev = (const float*)d_in[0];           // input spikes
    float* s4 = (float*)d_out;                             // first output half

    for (int l = 0; l < 4; ++l) {
        const float* a1 = (const float*)d_in[base[l] + 0];
        const float* a2 = (const float*)d_in[base[l] + 1];
        const float* bb = (const float*)d_in[base[l] + 2];
        const float* W  = (const float*)d_in[base[l] + 3];
        const float* bi = (const float*)d_in[base[l] + 4];

        int nIir = BATCHN * Kpad[l];
        iir_to_f16<<<(nIir + 255) / 256, 256, 0, stream>>>(
            s_prev, a1, a2, bb, xh, Fin[l], Kpad[l]);

        int nW = Mpad[l] * Kpad[l];
        pack_w<<<(nW + 255) / 256, 256, 0, stream>>>(
            W, Wh, Fout[l], Fin[l], Mpad[l], Kpad[l]);

        dim3 grid(T_PAD / 16, Mpad[l] / 16, BATCHN);
        gemm_wmma<<<grid, 32, 0, stream>>>(xh, Wh, bi, cur, Fout[l], Kpad[l]);

        float* sout = (l == 3) ? s4 : spk;
        int nS = BATCHN * Fout[l];
        spike_scan<<<(nS + 255) / 256, 256, 0, stream>>>(cur, sout, Fout[l]);

        s_prev = sout;
    }

    // final output filter into second half of d_out
    float* filt = s4 + (size_t)BATCHN * 300 * T_LEN;
    int nF = BATCHN * 300;
    out_filter<<<(nF + 255) / 256, 256, 0, stream>>>(s4, filt);
}
